// HGATLayer_42245298323504
// MI455X (gfx1250) — matrix operand order repaired and checked by reading, not compiled, not run on hardware
//
#include <hip/hip_runtime.h>

// ---------------------------------------------------------------------------
// CDNA5 (gfx1250) wave32 WMMA types
// ---------------------------------------------------------------------------
typedef __attribute__((ext_vector_type(16))) __bf16 v16bf;
typedef __attribute__((ext_vector_type(8)))  float  v8f;

union Frag {
  v16bf v;
  unsigned u[8];
  unsigned short s[16];
};

__device__ __forceinline__ unsigned short f2bf(float f) {
  unsigned u = __builtin_bit_cast(unsigned, f);
  unsigned r = u + 0x7FFFu + ((u >> 16) & 1u);   // round-to-nearest-even
  return (unsigned short)(r >> 16);
}

__device__ __forceinline__ unsigned pack2bf(float a, float b) {
  return (unsigned)f2bf(a) | ((unsigned)f2bf(b) << 16);
}

// ---------------------------------------------------------------------------
// Elementwise fp32 -> bf16 convert
// ---------------------------------------------------------------------------
__global__ void cvt_bf16(const float* __restrict__ in,
                         unsigned short* __restrict__ out, int n) {
  int i = blockIdx.x * blockDim.x + threadIdx.x;
  if (i < n) out[i] = f2bf(in[i]);
}

// Transpose 64x64 relation matrices (16 of them) while converting:
// out[m][e][d] = in[m][d][e]  (so B-fragment loads are contiguous over d)
__global__ void cvt_tr64(const float* __restrict__ in,
                         unsigned short* __restrict__ out, int n) {
  int i = blockIdx.x * blockDim.x + threadIdx.x;
  if (i < n) {
    int m = i >> 12, rem = i & 4095, d = rem >> 6, e = rem & 63;
    out[(m << 12) + (e << 6) + d] = f2bf(in[i]);
  }
}

// Pack int32 adjacency (…,1152) into 36 uint32 bit-words per row.
__global__ void pack_adj(const int* __restrict__ adj,
                         unsigned* __restrict__ bits, int total) {
  int i = blockIdx.x * blockDim.x + threadIdx.x;
  if (i >= total) return;
  int row = i / 36, w = i - row * 36;
  const int* p = adj + (size_t)row * 1152 + w * 32;
  unsigned v = 0;
  #pragma unroll
  for (int j = 0; j < 32; ++j) v |= (p[j] ? 1u : 0u) << j;
  bits[i] = v;
}

// ---------------------------------------------------------------------------
// Generic bf16 WMMA GEMM:  C[M,N] = A[M,K] * Bt[N,K]^T   (Bt row-major N x K)
// MODE 0: K=256,N=256; out bf16 scattered to head-major [b][h][n][64]
// MODE 1: K=64, N=64;  batched over blockIdx.y = b*4+h; out bf16 [z][n][64]
// MODE 2: K=256,N=256; out fp32 row-major (final output)
// MODE 3: K=64, N=64;  batched; out bf16 TRANSPOSED [z][64][nrows]
// block = 128 threads = 4 waves; each block computes 16 rows x N cols.
// ---------------------------------------------------------------------------
template <int MODE>
__global__ __launch_bounds__(128)
void gemm_bf16(const unsigned short* __restrict__ A,
               const unsigned short* __restrict__ Bt,
               void* __restrict__ Out, int Kdim, int nrows) {
  constexpr int NTILES = (MODE == 1 || MODE == 3) ? 1 : 4;
  const int lane = threadIdx.x & 31;
  const int wv   = threadIdx.x >> 5;
  const int colL = lane & 15;
  const int hlf  = lane >> 4;
  const int m0   = blockIdx.x * 16;

  const unsigned short* Ab = A;
  const unsigned short* Bb = Bt;
  unsigned short* Ob16 = (unsigned short*)Out;
  float* Of32 = (float*)Out;
  if (MODE == 1 || MODE == 3) {
    const int z = blockIdx.y;                 // z = b*4 + h
    Ab   += (size_t)z * nrows * 64;
    Bb   += (size_t)(z & 3) * 64 * 64;
    Ob16 += (size_t)z * nrows * 64;
  }

  const v8f zf = {0.f, 0.f, 0.f, 0.f, 0.f, 0.f, 0.f, 0.f};
  v8f c[NTILES];
  #pragma unroll
  for (int t = 0; t < NTILES; ++t) c[t] = zf;

  const size_t arow = (size_t)(m0 + colL) * Kdim;
  for (int kt = 0; kt < Kdim; kt += 32) {
    Frag a;                                   // A 16x32: lanes 0-15 = rows
    #pragma unroll
    for (int jj = 0; jj < 8; ++jj) {
      const int kk = kt + hlf * 16 + jj * 2;
      a.u[jj] = *(const unsigned*)(Ab + arow + kk);
    }
    #pragma unroll
    for (int t = 0; t < NTILES; ++t) {
      const int n0 = (NTILES == 1) ? wv * 16 : (wv + t * 4) * 16;
      Frag b;                                 // B 32x16: lanes 0-15 = cols
      #pragma unroll
      for (int jj = 0; jj < 8; ++jj) {
        const int kk = kt + hlf * 16 + jj * 2;
        b.u[jj] = *(const unsigned*)(Bb + (size_t)(n0 + colL) * Kdim + kk);
      }
      c[t] = __builtin_amdgcn_wmma_f32_16x16x32_bf16(
          false, a.v, false, b.v, (short)0, c[t], false, false);
    }
  }

  #pragma unroll
  for (int t = 0; t < NTILES; ++t) {
    const int n0  = (NTILES == 1) ? wv * 16 : (wv + t * 4) * 16;
    const int col = n0 + colL;
    if (MODE == 3) {
      // transposed store: out[col][row], 2 row-adjacent bf16 per dword
      #pragma unroll
      for (int r = 0; r < 8; r += 2) {
        const int row = m0 + r + 8 * hlf;
        *(unsigned*)(Ob16 + (size_t)col * nrows + row) =
            pack2bf(c[0][r], c[0][r + 1]);
      }
    } else {
      #pragma unroll
      for (int r = 0; r < 8; ++r) {
        const int row = m0 + r + 8 * hlf;
        const float vv = c[t][r];
        if (MODE == 0) {
          const int bi = row / nrows, rr = row % nrows;
          const int hh = col >> 6, ee = col & 63;
          Ob16[(((size_t)(bi * 4 + hh) * nrows + rr) << 6) + ee] = f2bf(vv);
        } else if (MODE == 1) {
          Ob16[(size_t)row * 64 + col] = f2bf(vv);
        } else {
          Of32[(size_t)row * 256 + col] = vv;
        }
      }
    }
  }
}

// ---------------------------------------------------------------------------
// Flash-style masked softmax attention, computed TRANSPOSED.
// One wave per (batch b, head h, 16-query tile). For each 32-key super-tile:
//   S^T = KA_tile(16k x 64d) x Q^T(64d x 16q)   (two 16-key sub-tiles)
// In S^T the lane = query, VGPR = key, so per-query softmax stats are
// per-lane scalars: the 32-key reduction is 15 in-lane VALU ops plus a
// single shfl_xor(16).  rep^T = V^T x P^T accumulates with lane = query,
// VGPR = feature.  P^T goes through a per-wave LDS slab (paired dwords).
// Writes relu(rep) bf16 into rep[b][q][h*64+e].
// ---------------------------------------------------------------------------
__global__ __launch_bounds__(128)
void attn_flash(const unsigned short* __restrict__ Q,    // [b][4][nq][64]
                const unsigned short* __restrict__ KAs,  // [b][4][1024][64]
                const unsigned short* __restrict__ KAo,  // [b][4][128][64]
                const unsigned short* __restrict__ VsT,  // [b][4][64][1024]
                const unsigned short* __restrict__ VoT,  // [b][4][64][128]
                const unsigned* __restrict__ adjbits,    // [b*nq][36]
                const float* __restrict__ pri,
                int priSelf, int priOth,
                unsigned short* __restrict__ rep,        // [b][nq][256]
                int nq) {
  __shared__ unsigned short Pst[4][16][32];              // 1 KB per wave
  const int lane = threadIdx.x & 31;
  const int h    = threadIdx.x >> 5;                     // wave = head
  const int colL = lane & 15;                            // query within tile
  const int hlf  = lane >> 4;
  const int q0   = blockIdx.x * 16;
  const int b    = blockIdx.y;

  const float scS = pri[priSelf + h] * 0.125f;           // pri / sqrt(64)
  const float scO = pri[priOth + h] * 0.125f;

  // Q as B fragments (B[d][q] = Q[q][d]; sequential-d layout)
  const unsigned short* Qr = Q + ((size_t)(b * 4 + h) * nq + q0 + colL) * 64;
  Frag qb0, qb1;
  #pragma unroll
  for (int jj = 0; jj < 8; ++jj) {
    const int dd = hlf * 16 + jj * 2;
    qb0.u[jj] = *(const unsigned*)(Qr + dd);
    qb1.u[jj] = *(const unsigned*)(Qr + 32 + dd);
  }

  const v8f zf = {0.f, 0.f, 0.f, 0.f, 0.f, 0.f, 0.f, 0.f};
  v8f acc[4];                                            // rep^T: 64 features
  #pragma unroll
  for (int t = 0; t < 4; ++t) acc[t] = zf;
  float mrun = -3.0e38f, lrun = 0.f;                     // per-lane (query)

  const unsigned short* KAsB = KAs + (size_t)(b * 4 + h) * 1024 * 64;
  const unsigned short* KAoB = KAo + (size_t)(b * 4 + h) * 128 * 64;
  const unsigned short* VsB  = VsT + (size_t)(b * 4 + h) * 64 * 1024;
  const unsigned short* VoB  = VoT + (size_t)(b * 4 + h) * 64 * 128;
  const unsigned* brow = adjbits + ((size_t)b * nq + q0 + colL) * 36;

  for (int kk0 = 0; kk0 < 1152; kk0 += 32) {             // 32-key super-tiles
    const bool self2 = (kk0 < 1024);                     // never straddles
    const float sc = self2 ? scS : scO;
    const unsigned wb = brow[kk0 >> 5];                  // this query's mask

    // ---- S^T for both 16-key sub-tiles (lane = query, VGPR = key) ----
    v8f s01[2];
    #pragma unroll
    for (int sub = 0; sub < 2; ++sub) {
      const int k0 = kk0 + sub * 16;
      const unsigned short* krow =
          (self2 ? (KAsB + (size_t)k0 * 64) : (KAoB + (size_t)(k0 - 1024) * 64))
          + (size_t)colL * 64;
      Frag ka0, ka1;                                     // A: rows = keys
      #pragma unroll
      for (int jj = 0; jj < 8; ++jj) {
        const int dd = hlf * 16 + jj * 2;
        ka0.u[jj] = *(const unsigned*)(krow + dd);
        ka1.u[jj] = *(const unsigned*)(krow + 32 + dd);
      }
      v8f s = __builtin_amdgcn_wmma_f32_16x16x32_bf16(
          false, ka0.v, false, qb0.v, (short)0, zf, false, false);
      s = __builtin_amdgcn_wmma_f32_16x16x32_bf16(
          false, ka1.v, false, qb1.v, (short)0, s, false, false);
      #pragma unroll
      for (int r = 0; r < 8; ++r) {
        const int bitpos = sub * 16 + 8 * hlf + r;       // key within word
        float sv = s[r] * sc;
        if (!((wb >> bitpos) & 1u)) sv = -3.0e38f;       // finfo.min mask
        s[r] = sv;
      }
      s01[sub] = s;
    }

    // ---- shared online-softmax step for the 32 keys ----
    float tmax = s01[0][0];
    #pragma unroll
    for (int r = 1; r < 8; ++r) tmax = fmaxf(tmax, s01[0][r]);
    #pragma unroll
    for (int r = 0; r < 8; ++r) tmax = fmaxf(tmax, s01[1][r]);
    tmax = fmaxf(tmax, __shfl_xor(tmax, 16, 32));        // other half's keys
    const float mnew = fmaxf(mrun, tmax);
    const float corr = __expf(mrun - mnew);
    mrun = mnew;

    float psum = 0.f;
    #pragma unroll
    for (int sub = 0; sub < 2; ++sub) {
      float p[8];
      #pragma unroll
      for (int r = 0; r < 8; ++r) {
        p[r] = __expf(s01[sub][r] - mnew);
        psum += p[r];
      }
      #pragma unroll
      for (int r = 0; r < 8; r += 2)                     // paired dword store
        *(unsigned*)&Pst[h][colL][sub * 16 + 8 * hlf + r] =
            pack2bf(p[r], p[r + 1]);
    }
    psum += __shfl_xor(psum, 16, 32);
    lrun = lrun * corr + psum;
    #pragma unroll
    for (int t = 0; t < 4; ++t)
      #pragma unroll
      for (int r = 0; r < 8; ++r) acc[t][r] = acc[t][r] * corr;

    asm volatile("" ::: "memory");                       // order LDS st->ld
    // ---- rep^T += V^T * P^T ----
    Frag pb;                                             // B: cols = queries
    #pragma unroll
    for (int jj = 0; jj < 8; ++jj)
      pb.u[jj] = *(const unsigned*)&Pst[h][colL][hlf * 16 + jj * 2];
    #pragma unroll
    for (int t = 0; t < 4; ++t) {
      const unsigned short* vrow =
          self2 ? (VsB + (size_t)(t * 16 + colL) * 1024 + kk0)
                : (VoB + (size_t)(t * 16 + colL) * 128 + (kk0 - 1024));
      Frag va;                                           // A: rows = features
      #pragma unroll
      for (int jj = 0; jj < 8; ++jj) {
        const int kk = hlf * 16 + jj * 2;
        va.u[jj] = *(const unsigned*)(vrow + kk);
      }
      acc[t] = __builtin_amdgcn_wmma_f32_16x16x32_bf16(
          false, va.v, false, pb.v, (short)0, acc[t], false, false);
    }
    asm volatile("" ::: "memory");
  }

  // ---- finalize: relu(acc / l), paired bf16 dword stores ----
  const float inv = 1.0f / lrun;
  unsigned short* repB =
      rep + (size_t)(b * nq + q0 + colL) * 256 + h * 64;
  #pragma unroll
  for (int t = 0; t < 4; ++t)
    #pragma unroll
    for (int r = 0; r < 8; r += 2) {
      const float v0 = fmaxf(acc[t][r] * inv, 0.f);
      const float v1 = fmaxf(acc[t][r + 1] * inv, 0.f);
      *(unsigned*)(repB + t * 16 + 8 * hlf + r) = pack2bf(v0, v1);
    }
}

// ---------------------------------------------------------------------------
// Host-side orchestration
// ---------------------------------------------------------------------------
extern "C" void kernel_launch(void* const* d_in, const int* in_sizes, int n_in,
                              void* d_out, int out_size, void* d_ws,
                              size_t ws_size, hipStream_t stream) {
  (void)in_sizes; (void)n_in; (void)out_size; (void)ws_size;
  const float* players    = (const float*)d_in[0];
  const float* teams      = (const float*)d_in[1];
  const int*   player_adj = (const int*)d_in[2];
  const int*   team_adj   = (const int*)d_in[3];
  const float* W[8];
  for (int i = 0; i < 8; ++i) W[i] = (const float*)d_in[4 + i];
  const float* rel_pri = (const float*)d_in[12];
  const float* rel_att = (const float*)d_in[13];
  const float* rel_msg = (const float*)d_in[14];

  char* base = (char*)d_ws;
  size_t off = 0;
  auto bump = [&](size_t elems, size_t esz) -> void* {
    off = (off + 255) & ~(size_t)255;
    void* p = base + off;
    off += elems * esz;
    return p;
  };
  const int NPel = 8 * 1024 * 256;   // 2,097,152
  const int NTel = 8 * 128 * 256;    //   262,144

  unsigned short* players_bf = (unsigned short*)bump(NPel, 2);
  unsigned short* teams_bf   = (unsigned short*)bump(NTel, 2);
  unsigned short* Wbf        = (unsigned short*)bump(8 * 65536, 2);
  unsigned short* attT       = (unsigned short*)bump(65536, 2);
  unsigned short* msgT       = (unsigned short*)bump(65536, 2);
  unsigned short* Qp   = (unsigned short*)bump(NPel, 2);
  unsigned short* Kp   = (unsigned short*)bump(NPel, 2);
  unsigned short* Vp   = (unsigned short*)bump(NPel, 2);
  unsigned short* Qt   = (unsigned short*)bump(NTel, 2);
  unsigned short* Kt   = (unsigned short*)bump(NTel, 2);
  unsigned short* Vt   = (unsigned short*)bump(NTel, 2);
  unsigned short* KAp2 = (unsigned short*)bump(NPel, 2);
  unsigned short* KAt3 = (unsigned short*)bump(NTel, 2);
  unsigned short* Mp0  = (unsigned short*)bump(NPel, 2);   // transposed
  unsigned short* Mp2  = (unsigned short*)bump(NPel, 2);   // transposed
  unsigned short* Mt1  = (unsigned short*)bump(NTel, 2);   // transposed
  unsigned short* Mt3  = (unsigned short*)bump(NTel, 2);   // transposed
  unsigned* pbits = (unsigned*)bump(8 * 1024 * 36, 4);
  unsigned* tbits = (unsigned*)bump(8 * 128 * 36, 4);
  // Aliases (lifetimes do not overlap, stream-ordered):
  unsigned short* KAp0 = players_bf;  // proj GEMMs precede rel GEMMs
  unsigned short* KAt1 = teams_bf;
  unsigned short* repP = Kp;          // K tensors dead before attention
  unsigned short* repT = Kt;

  auto cdiv = [](int a, int b) { return (a + b - 1) / b; };

  // ---- Stage 0: conversions / packing ----
  cvt_bf16<<<cdiv(NPel, 256), 256, 0, stream>>>(players, players_bf, NPel);
  cvt_bf16<<<cdiv(NTel, 256), 256, 0, stream>>>(teams, teams_bf, NTel);
  for (int i = 0; i < 8; ++i)
    cvt_bf16<<<cdiv(65536, 256), 256, 0, stream>>>(W[i], Wbf + i * 65536, 65536);
  cvt_tr64<<<cdiv(65536, 256), 256, 0, stream>>>(rel_att, attT, 65536);
  cvt_tr64<<<cdiv(65536, 256), 256, 0, stream>>>(rel_msg, msgT, 65536);
  pack_adj<<<cdiv(8 * 1024 * 36, 256), 256, 0, stream>>>(player_adj, pbits,
                                                         8 * 1024 * 36);
  pack_adj<<<cdiv(8 * 128 * 36, 256), 256, 0, stream>>>(team_adj, tbits,
                                                        8 * 128 * 36);

  // ---- Stage 1: projections -> head-major bf16 ----
  gemm_bf16<0><<<dim3(512), 128, 0, stream>>>(players_bf, Wbf + 0 * 65536, Qp, 256, 1024);
  gemm_bf16<0><<<dim3(512), 128, 0, stream>>>(players_bf, Wbf + 2 * 65536, Kp, 256, 1024);
  gemm_bf16<0><<<dim3(512), 128, 0, stream>>>(players_bf, Wbf + 4 * 65536, Vp, 256, 1024);
  gemm_bf16<0><<<dim3(64),  128, 0, stream>>>(teams_bf, Wbf + 1 * 65536, Qt, 256, 128);
  gemm_bf16<0><<<dim3(64),  128, 0, stream>>>(teams_bf, Wbf + 3 * 65536, Kt, 256, 128);
  gemm_bf16<0><<<dim3(64),  128, 0, stream>>>(teams_bf, Wbf + 5 * 65536, Vt, 256, 128);

  // ---- Stage 2: per-head relation transforms (batched over b*4+h) ----
  gemm_bf16<1><<<dim3(64, 32), 128, 0, stream>>>(Kp, attT + 0 * 16384, KAp0, 64, 1024);
  gemm_bf16<1><<<dim3(64, 32), 128, 0, stream>>>(Kp, attT + 2 * 16384, KAp2, 64, 1024);
  gemm_bf16<1><<<dim3(8, 32),  128, 0, stream>>>(Kt, attT + 1 * 16384, KAt1, 64, 128);
  gemm_bf16<1><<<dim3(8, 32),  128, 0, stream>>>(Kt, attT + 3 * 16384, KAt3, 64, 128);
  // message tensors stored transposed ([z][64][n]) for the attention kernel
  gemm_bf16<3><<<dim3(64, 32), 128, 0, stream>>>(Vp, msgT + 0 * 16384, Mp0, 64, 1024);
  gemm_bf16<3><<<dim3(64, 32), 128, 0, stream>>>(Vp, msgT + 2 * 16384, Mp2, 64, 1024);
  gemm_bf16<3><<<dim3(8, 32),  128, 0, stream>>>(Vt, msgT + 1 * 16384, Mt1, 64, 128);
  gemm_bf16<3><<<dim3(8, 32),  128, 0, stream>>>(Vt, msgT + 3 * 16384, Mt3, 64, 128);

  // ---- Stage 3: flash attention (player branch, team branch) ----
  // rel_pri offsets: [0,0,0,:]=0  [0,1,1,:]=12  [1,2,0,:]=40  [1,3,1,:]=60
  attn_flash<<<dim3(64, 8), 128, 0, stream>>>(Qp, KAp0, KAt1, Mp0, Mt1, pbits,
                                              rel_pri, 0, 12, repP, 1024);
  attn_flash<<<dim3(8, 8), 128, 0, stream>>>(Qt, KAp2, KAt3, Mp2, Mt3, tbits,
                                             rel_pri, 40, 60, repT, 128);

  // ---- Stage 4: output projections -> fp32 d_out ----
  gemm_bf16<2><<<dim3(512), 128, 0, stream>>>(repP, Wbf + 6 * 65536, d_out, 256, 1024);
  gemm_bf16<2><<<dim3(64), 128, 0, stream>>>(
      repT, Wbf + 7 * 65536, (void*)((float*)d_out + 8 * 1024 * 256), 256, 128);
}
